// MSDeformAttn_90383291777050
// MI455X (gfx1250) — matrix operand order repaired and checked by reading, not compiled, not run on hardware
//
#include <hip/hip_runtime.h>
#include <hip/hip_bf16.h>

#define LQ_     13294
#define LEN_IN_ 13294
#define B_      2
#define NH      8
#define DH      32
#define KDIM    256

typedef __attribute__((ext_vector_type(16))) __bf16       bf16x16;
typedef __attribute__((ext_vector_type(8)))  __bf16       bf16x8;
typedef __attribute__((ext_vector_type(8)))  float        floatx8;
typedef __attribute__((ext_vector_type(4)))  unsigned int uint32x4;
typedef __attribute__((ext_vector_type(8)))  int          int32x8;
typedef __attribute__((ext_vector_type(4)))  int          int32x4;

// ---------------------------------------------------------------------------
// Pack: W (K x N, f32 row-major) -> Wt (N x K, bf16) so B-fragments of the
// WMMA are contiguous 32B runs along K.
// ---------------------------------------------------------------------------
__global__ __launch_bounds__(256)
void pack_wt(const float* __restrict__ W, __bf16* __restrict__ Wt, int K, int N) {
    int i = blockIdx.x * 256 + threadIdx.x;
    if (i >= K * N) return;
    int n = i % N, k = i / N;
    Wt[(size_t)n * K + k] = (__bf16)W[i];
}

// ---------------------------------------------------------------------------
// GEMM: C[M x N] = A[M x 256] * W[256 x N] + bias.
//  - A f32 (converted to bf16 on the fly), W pre-packed bf16 (N x K),
//    f32 accumulate via v_wmma_f32_16x16x32_bf16.
//  - Block = 8 waves = 8 row-tiles sharing ONE 64-column quad. The block's
//    64x256 bf16 weight strip (32KB) is staged into LDS once by the Tensor
//    Data Mover (tensor_load_to_lds, TENSORcnt), then every wave reads its
//    B fragments from LDS.
//  - SOFTMAX16: fused per-16-column softmax (attention weights) using
//    16-lane __shfl_xor reductions (wave32; masks {1,2,4,8} stay in-half).
// ---------------------------------------------------------------------------
#define STRIP_BYTES  (64 * KDIM * 2)          // 32768 B
#define STRIP_E8     (STRIP_BYTES / 8)        // 4096 x 8B elements

template<int N, bool SOFTMAX16>
__global__ __launch_bounds__(256)
void gemm_wmma_bf16(const float* __restrict__ A, const __bf16* __restrict__ Wt,
                    const float* __restrict__ bias, float* __restrict__ C, int M) {
    constexpr int K = KDIM;
    constexpr int colQuads = N / 64;
    const int rowTiles = (M + 15) >> 4;

    const int quad     = blockIdx.x % colQuads;
    const int rowBase  = (blockIdx.x / colQuads) * 8;      // first row-tile of block
    const int waveIn   = threadIdx.x >> 5;
    const int rowTile  = rowBase + waveIn;
    const int tileN    = quad << 6;
    const int lane     = threadIdx.x & 31;
    const int lrow     = lane & 15;
    const int khalf    = lane >> 4;

    __shared__ __attribute__((aligned(32))) __bf16 ldsW[64 * KDIM];   // 32 KB strip

    // ---- TDM: wave 0 DMAs this block's weight strip (64 cols x 256 K, bf16)
    // from global into LDS as a 1-D tensor of 4096 8-byte elements.
    if (waveIn == 0) {
        unsigned long long ga =
            (unsigned long long)(uintptr_t)(Wt + (size_t)tileN * K);
        unsigned int laddr =
            (unsigned int)(unsigned long long)
                (__attribute__((address_space(3))) __bf16*)&ldsW[0];

        uint32x4 g0;
        g0[0] = 1u;                                       // count=1 (valid user D#)
        g0[1] = laddr;                                    // lds_addr (bytes)
        g0[2] = (unsigned int)ga;                         // global_addr[31:0]
        g0[3] = (unsigned int)(ga >> 32) | (2u << 30);    // global_addr[56:32] | type=2

        int32x8 g1;
        g1[0] = (int)(3u << 16);                          // data_size=8B; no multicast
        g1[1] = (int)((STRIP_E8 & 0xFFFFu) << 16);        // tensor_dim0[15:0]
        g1[2] = (int)((STRIP_E8 >> 16) | (1u << 16));     // tensor_dim0[31:16], tensor_dim1=1
        g1[3] = (int)((STRIP_E8 & 0xFFFFu) << 16);        // tile_dim0 = 4096
        g1[4] = 0;                                        // tile_dim1/2 unused
        g1[5] = (int)STRIP_E8;                            // tensor_dim0_stride
        g1[6] = 0;
        g1[7] = 0;

        int32x4 z4 = {0, 0, 0, 0};                        // groups 2/3 disabled (<=2D)
        int32x8 z8 = {0, 0, 0, 0, 0, 0, 0, 0};
        __builtin_amdgcn_tensor_load_to_lds(g0, g1, z4, z4, z8, 0);
        __builtin_amdgcn_s_wait_tensorcnt(0);             // strip resident in LDS
    }
    __syncthreads();

    if (rowTile < rowTiles) {
        const int tileM = rowTile << 4;
        int arow = tileM + lrow;
        if (arow >= M) arow = M - 1;                      // clamp edge rows
        const float* Ap = A + (size_t)arow * K;

        floatx8 acc[4] = {floatx8{}, floatx8{}, floatx8{}, floatx8{}};

#pragma unroll
        for (int k0 = 0; k0 < K; k0 += 32) {
            // A fragment (contiguous, matches B): e -> K = k0 + khalf*16 + e
            const float4* a0 = (const float4*)(Ap + k0 + khalf * 16);
            const float4* a1 = (const float4*)(Ap + k0 + khalf * 16 + 8);
            float4 v0 = a0[0], v1 = a0[1], v2 = a1[0], v3 = a1[1];
            bf16x16 af;
            af[0]  = (__bf16)v0.x; af[1]  = (__bf16)v0.y; af[2]  = (__bf16)v0.z; af[3]  = (__bf16)v0.w;
            af[4]  = (__bf16)v1.x; af[5]  = (__bf16)v1.y; af[6]  = (__bf16)v1.z; af[7]  = (__bf16)v1.w;
            af[8]  = (__bf16)v2.x; af[9]  = (__bf16)v2.y; af[10] = (__bf16)v2.z; af[11] = (__bf16)v2.w;
            af[12] = (__bf16)v3.x; af[13] = (__bf16)v3.y; af[14] = (__bf16)v3.z; af[15] = (__bf16)v3.w;

            // B fragment from LDS (contiguous, matches A): e -> K = k0 + khalf*16 + e
            const int kb = k0 + khalf * 16;
#pragma unroll
            for (int j = 0; j < 4; ++j) {
                bf16x16 bf = *(const bf16x16*)&ldsW[(size_t)(j * 16 + lrow) * K + kb];
                acc[j] = __builtin_amdgcn_wmma_f32_16x16x32_bf16(
                    false, af, false, bf, (short)0, acc[j], false, false);
            }
        }

        const int rbase = tileM + (khalf << 3);
#pragma unroll
        for (int j = 0; j < 4; ++j) {
            const int col = tileN + j * 16 + lrow;
            const float bc = bias[col];
#pragma unroll
            for (int r = 0; r < 8; ++r) {
                float v = acc[j][r] + bc;
                if constexpr (SOFTMAX16) {
                    float mx = v;
                    mx = fmaxf(mx, __shfl_xor(mx, 1, 32));
                    mx = fmaxf(mx, __shfl_xor(mx, 2, 32));
                    mx = fmaxf(mx, __shfl_xor(mx, 4, 32));
                    mx = fmaxf(mx, __shfl_xor(mx, 8, 32));
                    float e = __expf(v - mx);
                    float s = e;
                    s += __shfl_xor(s, 1, 32);
                    s += __shfl_xor(s, 2, 32);
                    s += __shfl_xor(s, 4, 32);
                    s += __shfl_xor(s, 8, 32);
                    v = e / s;
                }
                const int row = rbase + r;
                if (row < M) C[(size_t)row * N + col] = v;
            }
        }
    }
}

// ---------------------------------------------------------------------------
// Deformable sampling: one wave per (b, q, head); lane = channel d (0..31)
// so every gather of value[.., h, :] is a fully coalesced 128B transaction.
// value (27MB) is L2-resident on MI455X (192MB L2), so the ~1.7GB of gather
// traffic never leaves L2.
// ---------------------------------------------------------------------------
__global__ __launch_bounds__(256)
void msda_sample(const float* __restrict__ value, const float* __restrict__ off,
                 const float* __restrict__ aw, const float* __restrict__ refp,
                 float* __restrict__ outp) {
    const int total = B_ * LQ_ * NH;
    const int wave = blockIdx.x * 8 + (threadIdx.x >> 5);
    if (wave >= total) return;
    const int h  = wave & 7;
    const int bq = wave >> 3;                     // b*LQ + q
    const int b  = bq / LQ_;
    const int lane = threadIdx.x & 31;            // = d

    const float* awp  = aw  + (size_t)bq * 128 + h * 16;
    const float* offp = off + (size_t)bq * 256 + h * 32;
    const float* refq = refp + (size_t)bq * 8;
    const float* vbase = value + (size_t)b * LEN_IN_ * 256 + h * 32 + lane;

    const int HH[4] = {100, 50, 25, 13};
    const int WW[4] = {100, 50, 25, 13};
    const int ST[4] = {0, 10000, 12500, 13125};

    float acc = 0.0f;
#pragma unroll
    for (int lvl = 0; lvl < 4; ++lvl) {
        const int   Hi = HH[lvl], Wi = WW[lvl], s = ST[lvl];
        const float Hf = (float)Hi, Wf = (float)Wi;
        const float rx = refq[lvl * 2 + 0];
        const float ry = refq[lvl * 2 + 1];
#pragma unroll
        for (int p = 0; p < 4; ++p) {
            const float ox = offp[(lvl * 4 + p) * 2 + 0];
            const float oy = offp[(lvl * 4 + p) * 2 + 1];
            const float a  = awp[lvl * 4 + p];
            // loc = ref + off/[W,H];  x = loc.x*W - 0.5 = rx*W + ox - 0.5
            const float x = rx * Wf + ox - 0.5f;
            const float y = ry * Hf + oy - 0.5f;
            const float x0f = floorf(x), y0f = floorf(y);
            const int x0 = (int)x0f, y0 = (int)y0f;
            const float lx = x - x0f, ly = y - y0f;
            const float w00 = (1.0f - lx) * (1.0f - ly);
            const float w10 = lx * (1.0f - ly);
            const float w01 = (1.0f - lx) * ly;
            const float w11 = lx * ly;

            auto corner = [&](int xi, int yi, float w) -> float {
                const bool valid = (xi >= 0) && (xi < Wi) && (yi >= 0) && (yi < Hi);
                int xc = xi < 0 ? 0 : (xi > Wi - 1 ? Wi - 1 : xi);
                int yc = yi < 0 ? 0 : (yi > Hi - 1 ? Hi - 1 : yi);
                const float v = vbase[(size_t)(s + yc * Wi + xc) * 256];
                return valid ? w * v : 0.0f;
            };
            acc += a * (corner(x0, y0, w00) + corner(x0 + 1, y0, w10) +
                        corner(x0, y0 + 1, w01) + corner(x0 + 1, y0 + 1, w11));
        }
    }
    outp[(size_t)bq * 256 + h * 32 + lane] = acc;
}

// ---------------------------------------------------------------------------
extern "C" void kernel_launch(void* const* d_in, const int* in_sizes, int n_in,
                              void* d_out, int out_size, void* d_ws, size_t ws_size,
                              hipStream_t stream) {
    const float* query  = (const float*)d_in[0];
    const float* refp   = (const float*)d_in[1];
    const float* inflat = (const float*)d_in[2];
    // d_in[3]=spatial_shapes, d_in[4]=level_start_index: compile-time constants here
    const float* W_off  = (const float*)d_in[5];
    const float* b_off  = (const float*)d_in[6];
    const float* W_attn = (const float*)d_in[7];
    const float* b_attn = (const float*)d_in[8];
    const float* W_val  = (const float*)d_in[9];
    const float* b_val  = (const float*)d_in[10];
    const float* W_out  = (const float*)d_in[11];
    const float* b_out  = (const float*)d_in[12];
    float* out = (float*)d_out;

    const size_t M = (size_t)B_ * LQ_;            // 26588 rows
    char* p = (char*)d_ws;
    float*  val   = (float*)p;  p += M * 256 * sizeof(float);
    float*  off   = (float*)p;  p += M * 256 * sizeof(float);
    float*  aw    = (float*)p;  p += M * 128 * sizeof(float);
    float*  outp  = (float*)p;  p += M * 256 * sizeof(float);
    __bf16* WtV   = (__bf16*)p; p += 256 * 256 * sizeof(__bf16);
    __bf16* WtO   = (__bf16*)p; p += 256 * 256 * sizeof(__bf16);
    __bf16* WtA   = (__bf16*)p; p += 128 * 256 * sizeof(__bf16);
    __bf16* WtU   = (__bf16*)p; p += 256 * 256 * sizeof(__bf16);

    // 1) pack weights -> bf16 transposed (N x K)
    pack_wt<<<(256 * 256 + 255) / 256, 256, 0, stream>>>(W_val,  WtV, 256, 256);
    pack_wt<<<(256 * 256 + 255) / 256, 256, 0, stream>>>(W_off,  WtO, 256, 256);
    pack_wt<<<(256 * 128 + 255) / 256, 256, 0, stream>>>(W_attn, WtA, 256, 128);
    pack_wt<<<(256 * 256 + 255) / 256, 256, 0, stream>>>(W_out,  WtU, 256, 256);

    const int rowTiles  = (int)((M + 15) / 16);        // 1662
    const int rowBlocks = (rowTiles + 7) / 8;          // 208 (8 row-tiles / block)
    const int blocks256 = rowBlocks * 4;               // N=256 -> 4 col-quads
    const int blocks128 = rowBlocks * 2;               // N=128 -> 2 col-quads

    // 2) value / offsets / attention-weight GEMMs (softmax fused into attn)
    gemm_wmma_bf16<256, false><<<blocks256, 256, 0, stream>>>(inflat, WtV, b_val,  val, (int)M);
    gemm_wmma_bf16<256, false><<<blocks256, 256, 0, stream>>>(query,  WtO, b_off,  off, (int)M);
    gemm_wmma_bf16<128, true ><<<blocks128, 256, 0, stream>>>(query,  WtA, b_attn, aw,  (int)M);

    // 3) deformable bilinear sampling: one wave per (b,q,head)
    const int sblocks = (int)((M * NH + 7) / 8);
    msda_sample<<<sblocks, 256, 0, stream>>>(val, off, aw, refp, outp);

    // 4) output projection
    gemm_wmma_bf16<256, false><<<blocks256, 256, 0, stream>>>(outp, WtU, b_out, out, (int)M);
}
